// SpikingAssocBlock_55576876810896
// MI455X (gfx1250) — matrix operand order repaired and checked by reading, not compiled, not run on hardware
//
#include <hip/hip_runtime.h>
#include <hip/hip_bf16.h>

typedef _Float16 v16h __attribute__((ext_vector_type(16)));
typedef _Float16 v8h  __attribute__((ext_vector_type(8)));
typedef _Float16 v4h  __attribute__((ext_vector_type(4)));
typedef float    v8f  __attribute__((ext_vector_type(8)));
typedef int      v4i  __attribute__((ext_vector_type(4)));

constexpr int Tdim = 2048, Bdim = 8, Hdim = 1024;
constexpr int Mdim = Tdim * Bdim;     // 16384 rows (t*B+b)
constexpr int Kdim = Hdim;            // 1024
constexpr int Ndim = Hdim;            // 1024

// GEMM tiling: block = 128(M) x 256(N), K-step 32, double-buffered LDS.
// 8 waves (256 threads, wave32) arranged 2x4; each wave owns 64x64 = 4x4 WMMA tiles.
constexpr int BM = 128, BN = 256, BK = 32;
constexpr int LDSL = BK + 8;          // 40 halfs = 80B row stride: 16B-aligned, conflict-free b128
constexpr int BUFH = (BM + BN) * LDSL; // halfs per LDS buffer (A then B)

constexpr int NCH    = Bdim * Hdim;   // 8192 channels
constexpr int CHUNKS = 32;
constexpr int CLEN   = Tdim / CHUNKS; // 64

__device__ __forceinline__ float sigmoidf_(float x) {
    return 1.0f / (1.0f + expf(-x));
}

// ---- CDNA5 async global->LDS copy (ASYNCcnt path), 16B per lane ----
__device__ __forceinline__ void async_ld16(const _Float16* gsrc, _Float16* ldst) {
#if __has_builtin(__builtin_amdgcn_global_load_async_to_lds_b128)
    __builtin_amdgcn_global_load_async_to_lds_b128(
        (__attribute__((address_space(1))) v4i*)(v4i*)(uintptr_t)gsrc,
        (__attribute__((address_space(3))) v4i*)(v4i*)ldst, 0, 0);
#else
    unsigned loff = (unsigned)(uintptr_t)(__attribute__((address_space(3))) void*)(void*)ldst;
    unsigned long long ga = (unsigned long long)(uintptr_t)gsrc;
    asm volatile("global_load_async_to_lds_b128 %0, %1, off"
                 :: "v"(loff), "v"(ga) : "memory");
#endif
}

template<int N>
__device__ __forceinline__ void wait_asynccnt() {
#if __has_builtin(__builtin_amdgcn_s_wait_asynccnt)
    __builtin_amdgcn_s_wait_asynccnt(N);
#else
    asm volatile("s_wait_asynccnt %0" :: "n"(N) : "memory");
#endif
}

// ---- f32 -> f16 conversion pass (x, W_q, W_fc), vectorized float4 -> half4 ----
__global__ __launch_bounds__(256)
void cvt_f32_f16(const float* __restrict__ in, _Float16* __restrict__ out, int n4) {
    int i = blockIdx.x * 256 + threadIdx.x;
    if (i >= n4) return;
    const float4 v = reinterpret_cast<const float4*>(in)[i];
    v4h h = { (_Float16)v.x, (_Float16)v.y, (_Float16)v.z, (_Float16)v.w };
    reinterpret_cast<v4h*>(out)[i] = h;
}

// out[m,n] = act( sum_k A[m,k] * W[n,k] + bias[n] ), A/W f16 row-major with ld=Kdim.
template<bool RELU>
__global__ __launch_bounds__(256)
void gemm_wmma(const _Float16* __restrict__ A, const _Float16* __restrict__ W,
               const float* __restrict__ bias, float* __restrict__ out)
{
    __shared__ _Float16 lds[2 * BUFH];   // 60 KB: [buf][A 128x40 | B 256x40]

    const int tid  = threadIdx.x;
    const int lane = tid & 31;
    const int wid  = tid >> 5;
    const int wm0  = (wid >> 2) * 64;
    const int wn0  = (wid & 3)  * 64;
    const int row0 = blockIdx.y * BM;
    const int col0 = blockIdx.x * BN;
    const int lr   = lane & 15;          // tile row (A) / tile col (B,C)
    const int lhi  = lane >> 4;          // 0 or 1

    // Issue 6 async B128 copies per wave for one K-step: A 512 chunks + B 1024 chunks of 16B.
    auto stage = [&](int buf, int kk) {
        _Float16* baseA = &lds[buf * BUFH];
        #pragma unroll
        for (int i = 0; i < 2; ++i) {
            int c = tid + i * 256;           // 512 chunks: 128 rows x 4
            int r = c >> 2, q = (c & 3) * 8; // q in halfs (8 halfs = 16B)
            async_ld16(A + (size_t)(row0 + r) * Kdim + kk + q, baseA + r * LDSL + q);
        }
        _Float16* baseB = baseA + BM * LDSL;
        #pragma unroll
        for (int i = 0; i < 4; ++i) {
            int c = tid + i * 256;           // 1024 chunks: 256 rows x 4
            int r = c >> 2, q = (c & 3) * 8;
            async_ld16(W + (size_t)(col0 + r) * Kdim + kk + q, baseB + r * LDSL + q);
        }
    };

    v8f acc[4][4] = {};

    constexpr int NS = Kdim / BK;        // 32 K-steps
    stage(0, 0);                         // prologue
    for (int s = 0; s < NS; ++s) {
        const int cur = s & 1;
        if (s + 1 < NS) {
            stage(cur ^ 1, (s + 1) * BK);   // overlap next tile's HBM/L2 traffic with WMMAs
            wait_asynccnt<6>();             // in-order completion => cur's 6 copies landed
        } else {
            wait_asynccnt<0>();
        }
        __syncthreads();                    // all waves' copies visible

        const _Float16* baseA = &lds[cur * BUFH];
        const _Float16* baseB = baseA + BM * LDSL;
        v16h af[4], bf[4];
        #pragma unroll
        for (int mi = 0; mi < 4; ++mi) {
            // A 16x32 layout: lane<16 -> M=lane, K {0..7,16..23}; lane>=16 -> K {8..15,24..31}
            const _Float16* p = baseA + (wm0 + mi * 16 + lr) * LDSL + lhi * 8;
            v8h lo = *reinterpret_cast<const v8h*>(p);
            v8h hi = *reinterpret_cast<const v8h*>(p + 16);
            af[mi] = __builtin_shufflevector(lo, hi, 0,1,2,3,4,5,6,7,8,9,10,11,12,13,14,15);
        }
        #pragma unroll
        for (int ni = 0; ni < 4; ++ni) {
            // B 32x16 layout: lane<16 -> N=lane, K {0..7,16..23}; lane>=16 -> K {8..15,24..31}
            const _Float16* p = baseB + (wn0 + ni * 16 + lr) * LDSL + lhi * 8;
            v8h lo = *reinterpret_cast<const v8h*>(p);
            v8h hi = *reinterpret_cast<const v8h*>(p + 16);
            bf[ni] = __builtin_shufflevector(lo, hi, 0,1,2,3,4,5,6,7,8,9,10,11,12,13,14,15);
        }
        #pragma unroll
        for (int mi = 0; mi < 4; ++mi)
            #pragma unroll
            for (int ni = 0; ni < 4; ++ni)
                acc[mi][ni] = __builtin_amdgcn_wmma_f32_16x16x32_f16(
                    false, af[mi], false, bf[ni], (short)0, acc[mi][ni], false, false);

        __syncthreads();                    // all waves done reading cur -> safe to overwrite
    }

    // Epilogue: C layout VGPR r -> M = r + 8*lhi, N = lr.
    float bn[4];
    #pragma unroll
    for (int ni = 0; ni < 4; ++ni)
        bn[ni] = bias[col0 + wn0 + ni * 16 + lr];
    #pragma unroll
    for (int mi = 0; mi < 4; ++mi) {
        #pragma unroll
        for (int ni = 0; ni < 4; ++ni) {
            const int n = col0 + wn0 + ni * 16 + lr;
            #pragma unroll
            for (int r = 0; r < 8; ++r) {
                const int m = row0 + wm0 + mi * 16 + r + lhi * 8;
                float v = acc[mi][ni][r] + bn[ni];
                if (RELU) v = fmaxf(v, 0.0f);
                out[(size_t)m * Ndim + n] = v;
            }
        }
    }
}

// Phase A: local scan within each chunk of 64 timesteps (f32 in place) + chunk-last value.
__global__ __launch_bounds__(256)
void scan_local(float* __restrict__ qm, const float* __restrict__ beta_raw,
                float* __restrict__ carry)
{
    const int gid = blockIdx.x * 256 + threadIdx.x;   // 32 chunks * 8192 channels
    const int ch  = gid & (NCH - 1);
    const int c   = gid >> 13;
    const int h   = ch & (Hdim - 1);
    const float beta = sigmoidf_(beta_raw[h]);
    const size_t base = (size_t)c * CLEN * NCH + ch;  // (t*B+b)*H+h = t*NCH + ch
    float m = 0.0f;
    for (int tl = 0; tl < CLEN; ++tl) {
        const size_t o = base + (size_t)tl * NCH;
        m = fmaf(beta, m, qm[o]);
        qm[o] = m;
    }
    carry[(size_t)c * NCH + ch] = m;                  // chunk-local last L_c
}

// Phase B: per-channel carry chain: Carry_c = beta^64 * Carry_{c-1} + L_{c-1}.
__global__ __launch_bounds__(256)
void scan_carry(float* __restrict__ carry, const float* __restrict__ beta_raw)
{
    const int ch = blockIdx.x * 256 + threadIdx.x;    // 8192
    const int h  = ch & (Hdim - 1);
    const float beta = sigmoidf_(beta_raw[h]);
    float b64 = beta;
    #pragma unroll
    for (int i = 0; i < 6; ++i) b64 *= b64;           // beta^64
    float Carry = 0.0f;
    for (int c = 0; c < CHUNKS; ++c) {
        const size_t o = (size_t)c * NCH + ch;
        const float L = carry[o];
        carry[o] = Carry;
        Carry = fmaf(b64, Carry, L);
    }
}

// Phase C: m_t = local_m_t + beta^(tl+1) * Carry_c, emitted directly as f16 for GEMM2.
__global__ __launch_bounds__(256)
void scan_apply_cvt(const float* __restrict__ qm, const float* __restrict__ beta_raw,
                    const float* __restrict__ carry, _Float16* __restrict__ mh)
{
    const int gid = blockIdx.x * 256 + threadIdx.x;
    const int ch  = gid & (NCH - 1);
    const int c   = gid >> 13;
    const int h   = ch & (Hdim - 1);
    const float beta  = sigmoidf_(beta_raw[h]);
    const float Carry = carry[(size_t)c * NCH + ch];
    const size_t base = (size_t)c * CLEN * NCH + ch;
    float pc = beta * Carry;
    for (int tl = 0; tl < CLEN; ++tl) {
        const size_t o = base + (size_t)tl * NCH;
        mh[o] = (_Float16)(qm[o] + pc);
        pc *= beta;
    }
}

extern "C" void kernel_launch(void* const* d_in, const int* in_sizes, int n_in,
                              void* d_out, int out_size, void* d_ws, size_t ws_size,
                              hipStream_t stream)
{
    const float* x        = (const float*)d_in[0];   // (T,B,H)
    const float* W_q      = (const float*)d_in[1];   // (H,H)
    const float* b_q      = (const float*)d_in[2];   // (H,)
    const float* beta_raw = (const float*)d_in[3];   // (H,)
    const float* W_fc     = (const float*)d_in[4];   // (H,H)
    const float* b_fc     = (const float*)d_in[5];   // (H,)
    float* out = (float*)d_out;

    // Workspace layout
    float*    q     = (float*)d_ws;                          // 64 MB (q, scanned in place)
    float*    carry = q + (size_t)Mdim * Ndim;               // 1 MB
    _Float16* xh    = (_Float16*)(carry + CHUNKS * NCH);     // 32 MB
    _Float16* mh    = xh + (size_t)Mdim * Kdim;              // 32 MB
    _Float16* wqh   = mh + (size_t)Mdim * Kdim;              // 2 MB
    _Float16* wfh   = wqh + (size_t)Ndim * Kdim;             // 2 MB

    const dim3 gemm_grid(Ndim / BN, Mdim / BM);              // (4, 128)

    // 0) one-shot f32->f16 conversions (halves all GEMM read traffic, enables async copies)
    cvt_f32_f16<<<(Mdim * Kdim / 4) / 256, 256, 0, stream>>>(x,    xh,  Mdim * Kdim / 4);
    cvt_f32_f16<<<(Ndim * Kdim / 4) / 256, 256, 0, stream>>>(W_q,  wqh, Ndim * Kdim / 4);
    cvt_f32_f16<<<(Ndim * Kdim / 4) / 256, 256, 0, stream>>>(W_fc, wfh, Ndim * Kdim / 4);
    // 1) q = x @ W_q^T + b_q  (f32 accumulate + f32 output for the scan)
    gemm_wmma<false><<<gemm_grid, 256, 0, stream>>>(xh, wqh, b_q, q);
    // 2) leaky-integration scan over T (chunked associative scan)
    scan_local<<<(CHUNKS * NCH) / 256, 256, 0, stream>>>(q, beta_raw, carry);
    scan_carry<<<NCH / 256, 256, 0, stream>>>(carry, beta_raw);
    scan_apply_cvt<<<(CHUNKS * NCH) / 256, 256, 0, stream>>>(q, beta_raw, carry, mh);
    // 3) out = relu(m @ W_fc^T + b_fc)
    gemm_wmma<true><<<gemm_grid, 256, 0, stream>>>(mh, wfh, b_fc, out);
}
